// Cross_Attention_75084618269447
// MI455X (gfx1250) — matrix operand order repaired and checked by reading, not compiled, not run on hardware
//
#include <hip/hip_runtime.h>
#include <hip/hip_bf16.h>

// ---------------------------------------------------------------------------
// Cross-attention on gfx1250 via v_wmma_f32_16x16x32_bf16 (wave32).
//   Q = X1 Wq ; K = X2 Wk ; V = X2 Wv ; O = softmax(Q K^T) V
// Each wave owns a 16x64 output strip (4 accumulators, A-fragment reused 4x).
// The K-loop is a two-stage software pipeline: two fragment sets are kept in
// flight (~20 outstanding b128 loads) so global-load latency is hidden behind
// 8 WMMA issues per iteration.
// ---------------------------------------------------------------------------

typedef __attribute__((ext_vector_type(16))) __bf16 v16bf;
typedef __attribute__((ext_vector_type(8)))  __bf16 v8bf;
typedef __attribute__((ext_vector_type(8)))  float  v8f;

#define SEQ    4096
#define DMODEL 1024

union AFrag { v16bf v; v8bf h[2]; };

// A-matrix fragment (16x32 bf16, row-major, row stride lda elements).
// k order used here (contiguous, same as B): lane l (l<16) row M=l, K = [16h..16h+15],
// h = lane>>4.  Two 16-byte loads per lane.
__device__ __forceinline__ v16bf load_a_frag(const __bf16* A, int lda,
                                             int mBase, int kBase, int lane) {
  const int l = lane & 15, h = lane >> 4;
  const __bf16* p = A + (size_t)(mBase + l) * lda + kBase + h * 16;
  AFrag f;
  f.h[0] = *(const v8bf*)(p);
  f.h[1] = *(const v8bf*)(p + 8);
  return f.v;
}

// B-matrix fragment (32x16 bf16).  Bt holds B transposed row-major
// (Bt[n][k] = B[k][n]); lane l (col n=l&15, K-half h=lane>>4) does one
// contiguous 32-byte (32B-aligned) load.
__device__ __forceinline__ v16bf load_b_frag(const __bf16* Bt, int ldb,
                                             int nBase, int kBase, int lane) {
  const int l = lane & 15, h = lane >> 4;
  const __bf16* p = Bt + (size_t)(nBase + l) * ldb + kBase + h * 16;
  return *(const v16bf*)(p);
}

// One K-step fragment set: A tile + 4 B tiles covering a 16x64 strip.
struct FragSet { v16bf a, b0, b1, b2, b3; };

__device__ __forceinline__ FragSet load_set(const __bf16* A, int lda, int mBase,
                                            const __bf16* Bt, int ldb, int nBase,
                                            int kk, int lane) {
  FragSet f;
  f.a  = load_a_frag(A, lda, mBase, kk, lane);
  f.b0 = load_b_frag(Bt, ldb, nBase +  0, kk, lane);
  f.b1 = load_b_frag(Bt, ldb, nBase + 16, kk, lane);
  f.b2 = load_b_frag(Bt, ldb, nBase + 32, kk, lane);
  f.b3 = load_b_frag(Bt, ldb, nBase + 48, kk, lane);
  return f;
}

__device__ __forceinline__ void mma4(const FragSet& f,
                                     v8f& a0, v8f& a1, v8f& a2, v8f& a3) {
  a0 = __builtin_amdgcn_wmma_f32_16x16x32_bf16(false, f.a, false, f.b0, (short)0, a0, false, false);
  a1 = __builtin_amdgcn_wmma_f32_16x16x32_bf16(false, f.a, false, f.b1, (short)0, a1, false, false);
  a2 = __builtin_amdgcn_wmma_f32_16x16x32_bf16(false, f.a, false, f.b2, (short)0, a2, false, false);
  a3 = __builtin_amdgcn_wmma_f32_16x16x32_bf16(false, f.a, false, f.b3, (short)0, a3, false, false);
}

// Two-stage pipelined 16x64-strip GEMM over reduction length K (K % 64 == 0).
__device__ __forceinline__ void gemm_strip(const __bf16* A, int lda, int mBase,
                                           const __bf16* Bt, int ldb, int nBase,
                                           int K, int lane,
                                           v8f& a0, v8f& a1, v8f& a2, v8f& a3) {
  a0 = {}; a1 = {}; a2 = {}; a3 = {};
  FragSet f0 = load_set(A, lda, mBase, Bt, ldb, nBase, 0, lane);
  FragSet f1 = load_set(A, lda, mBase, Bt, ldb, nBase, 32, lane);
#pragma unroll 1
  for (int kk = 64; kk < K; kk += 64) {
    FragSet f2 = load_set(A, lda, mBase, Bt, ldb, nBase, kk, lane);       // prefetch
    mma4(f0, a0, a1, a2, a3);                                             // consume
    FragSet f3 = load_set(A, lda, mBase, Bt, ldb, nBase, kk + 32, lane);  // prefetch
    mma4(f1, a0, a1, a2, a3);                                             // consume
    f0 = f2; f1 = f3;
  }
  mma4(f0, a0, a1, a2, a3);  // epilogue
  mma4(f1, a0, a1, a2, a3);
}

// ---------------------------------------------------------------------------
// Kernel 0a: f32 -> bf16 elementwise
__global__ void k_cvt_bf16(const float* __restrict__ in,
                           __bf16* __restrict__ out, int n) {
  int i = blockIdx.x * blockDim.x + threadIdx.x;
  if (i < n) out[i] = (__bf16)in[i];
}

// Kernel 0b: f32 -> bf16 with transpose: out[c*rows + r] = in[r*cols + c]
__global__ void k_cvt_bf16_T(const float* __restrict__ in,
                             __bf16* __restrict__ out, int rows, int cols) {
  int i = blockIdx.x * blockDim.x + threadIdx.x;
  if (i < rows * cols) {
    int r = i / cols, c = i % cols;
    out[(size_t)c * rows + r] = (__bf16)in[(size_t)r * cols + c];
  }
}

// ---------------------------------------------------------------------------
// Kernel 1: projection GEMMs.  One wave = one 16x64 output strip.
//   which 0: Qb = X1b * Wq   (row-major bf16 out)
//   which 1: Kb = X2b * Wk   (row-major bf16 out)
//   which 2: Vt = (X2b * Wv)^T  (transposed bf16 out, [DMODEL][SEQ])
__global__ void k_proj(const __bf16* __restrict__ X1b,
                       const __bf16* __restrict__ X2b,
                       const __bf16* __restrict__ WqT,
                       const __bf16* __restrict__ WkT,
                       const __bf16* __restrict__ WvT,
                       __bf16* __restrict__ Qb,
                       __bf16* __restrict__ Kb,
                       __bf16* __restrict__ Vt) {
  const int lane   = threadIdx.x & 31;
  const int waveId = (blockIdx.x * blockDim.x + threadIdx.x) >> 5;

  const int tilesM  = SEQ / 16;      // 256
  const int tilesN4 = DMODEL / 64;   // 16 strips of 64
  const int tilesPer = tilesM * tilesN4;
  const int which = waveId / tilesPer;
  if (which >= 3) return;
  const int rem   = waveId % tilesPer;
  const int mBase = (rem / tilesN4) * 16;
  const int nBase = (rem % tilesN4) * 64;

  const __bf16* A  = (which == 0) ? X1b : X2b;
  const __bf16* Bt = (which == 0) ? WqT : (which == 1) ? WkT : WvT;

  v8f acc0, acc1, acc2, acc3;
  gemm_strip(A, DMODEL, mBase, Bt, DMODEL, nBase, DMODEL, lane,
             acc0, acc1, acc2, acc3);

  const int l = lane & 15, h = lane >> 4;
  v8f accs[4] = {acc0, acc1, acc2, acc3};
  if (which == 2) {
    // transposed store: Vt[nBase+16t+l][mBase + 8h + r], contiguous over r
#pragma unroll
    for (int t = 0; t < 4; ++t) {
      v8bf pv;
#pragma unroll
      for (int r = 0; r < 8; ++r) pv[r] = (__bf16)accs[t][r];
      *(v8bf*)(Vt + (size_t)(nBase + 16 * t + l) * SEQ + mBase + 8 * h) = pv;
    }
  } else {
    __bf16* O = (which == 0) ? Qb : Kb;
#pragma unroll
    for (int t = 0; t < 4; ++t)
#pragma unroll
      for (int r = 0; r < 8; ++r)
        O[(size_t)(mBase + r + 8 * h) * DMODEL + nBase + 16 * t + l] = (__bf16)accs[t][r];
  }
}

// ---------------------------------------------------------------------------
// Kernel 2: softmax.  One wave owns 16 query rows; processes 64 keys/step.
// Sweep 1: online (rowmax, rowsum).  Sweep 2: store P = exp(S-max)/sum (bf16).
__global__ void k_softmax(const __bf16* __restrict__ Qb,
                          const __bf16* __restrict__ Kb,
                          __bf16* __restrict__ P) {
  const int lane   = threadIdx.x & 31;
  const int waveId = (blockIdx.x * blockDim.x + threadIdx.x) >> 5;
  if (waveId >= SEQ / 16) return;
  const int mBase = waveId * 16;
  const int l = lane & 15, h = lane >> 4;

  float Mrun[8], Srun[8];
#pragma unroll
  for (int r = 0; r < 8; ++r) { Mrun[r] = -INFINITY; Srun[r] = 0.0f; }

  // ---- sweep 1: statistics over all keys, 64 at a time ----
  for (int nb = 0; nb < SEQ; nb += 64) {
    v8f s0, s1, s2, s3;
    // B[k][n] = K[nb+n][k] -> row-major Kb acts as "Bt"
    gemm_strip(Qb, DMODEL, mBase, Kb, DMODEL, nb, DMODEL, lane, s0, s1, s2, s3);
#pragma unroll
    for (int r = 0; r < 8; ++r) {
      float x0 = s0[r], x1 = s1[r], x2 = s2[r], x3 = s3[r];
      // local max over 4 tiles, then one butterfly over the 16-lane half
      float tmax = fmaxf(fmaxf(x0, x1), fmaxf(x2, x3));
#pragma unroll
      for (int off = 1; off < 16; off <<= 1)
        tmax = fmaxf(tmax, __shfl_xor(tmax, off, 32));
      float newM = fmaxf(Mrun[r], tmax);
      float ps = __expf(x0 - newM) + __expf(x1 - newM) +
                 __expf(x2 - newM) + __expf(x3 - newM);
#pragma unroll
      for (int off = 1; off < 16; off <<= 1)
        ps += __shfl_xor(ps, off, 32);
      Srun[r] = Srun[r] * __expf(Mrun[r] - newM) + ps;
      Mrun[r] = newM;
    }
  }

  float invS[8];
#pragma unroll
  for (int r = 0; r < 8; ++r) invS[r] = 1.0f / Srun[r];

  // ---- sweep 2: normalized probabilities ----
  for (int nb = 0; nb < SEQ; nb += 64) {
    v8f s0, s1, s2, s3;
    gemm_strip(Qb, DMODEL, mBase, Kb, DMODEL, nb, DMODEL, lane, s0, s1, s2, s3);
    v8f ss[4] = {s0, s1, s2, s3};
#pragma unroll
    for (int t = 0; t < 4; ++t)
#pragma unroll
      for (int r = 0; r < 8; ++r) {
        float p = __expf(ss[t][r] - Mrun[r]) * invS[r];
        P[(size_t)(mBase + r + 8 * h) * SEQ + nb + 16 * t + l] = (__bf16)p;
      }
  }
}

// ---------------------------------------------------------------------------
// Kernel 3: O = P * V.  One wave = 16x64 strip of the output; P-fragment
// loaded once per K-step, reused across 4 dV tiles (Vt gives row loads).
__global__ void k_out(const __bf16* __restrict__ P,
                      const __bf16* __restrict__ Vt,
                      float* __restrict__ out) {
  const int lane   = threadIdx.x & 31;
  const int waveId = (blockIdx.x * blockDim.x + threadIdx.x) >> 5;
  const int tilesN4 = DMODEL / 64;   // 16
  if (waveId >= (SEQ / 16) * tilesN4) return;
  const int mBase = (waveId / tilesN4) * 16;
  const int nBase = (waveId % tilesN4) * 64;

  v8f acc0, acc1, acc2, acc3;
  // Vt[n][k] = V[k][n] -> contiguous B-fragment rows
  gemm_strip(P, SEQ, mBase, Vt, SEQ, nBase, SEQ, lane, acc0, acc1, acc2, acc3);

  const int l = lane & 15, h = lane >> 4;
  v8f accs[4] = {acc0, acc1, acc2, acc3};
#pragma unroll
  for (int t = 0; t < 4; ++t)
#pragma unroll
    for (int r = 0; r < 8; ++r)
      out[(size_t)(mBase + r + 8 * h) * DMODEL + nBase + 16 * t + l] = accs[t][r];
}

// ---------------------------------------------------------------------------
extern "C" void kernel_launch(void* const* d_in, const int* in_sizes, int n_in,
                              void* d_out, int out_size, void* d_ws, size_t ws_size,
                              hipStream_t stream) {
  const float* x1 = (const float*)d_in[0];
  const float* x2 = (const float*)d_in[1];
  const float* wq = (const float*)d_in[2];
  const float* wk = (const float*)d_in[3];
  const float* wv = (const float*)d_in[4];
  float* out = (float*)d_out;

  // Workspace layout (bytes)
  char* ws = (char*)d_ws;
  const size_t SZ_X = (size_t)SEQ * DMODEL * sizeof(__bf16);     // 8 MB
  const size_t SZ_W = (size_t)DMODEL * DMODEL * sizeof(__bf16);  // 2 MB
  __bf16* X1b = (__bf16*)(ws);
  __bf16* X2b = (__bf16*)(ws + SZ_X);
  __bf16* WqT = (__bf16*)(ws + 2 * SZ_X);
  __bf16* WkT = (__bf16*)(ws + 2 * SZ_X + SZ_W);
  __bf16* WvT = (__bf16*)(ws + 2 * SZ_X + 2 * SZ_W);
  __bf16* Qb  = (__bf16*)(ws + 2 * SZ_X + 3 * SZ_W);
  __bf16* Kb  = (__bf16*)(ws + 3 * SZ_X + 3 * SZ_W);
  __bf16* Vt  = (__bf16*)(ws + 4 * SZ_X + 3 * SZ_W);
  __bf16* P   = (__bf16*)(ws + 5 * SZ_X + 3 * SZ_W);   // 32 MB
  (void)ws_size; (void)in_sizes; (void)n_in; (void)out_size;

  const int THREADS = 256;  // 8 wave32 per block

  // 0) precision conversion + weight transposes
  {
    int n = SEQ * DMODEL;
    k_cvt_bf16<<<(n + THREADS - 1) / THREADS, THREADS, 0, stream>>>(x1, X1b, n);
    k_cvt_bf16<<<(n + THREADS - 1) / THREADS, THREADS, 0, stream>>>(x2, X2b, n);
    int nw = DMODEL * DMODEL;
    k_cvt_bf16_T<<<(nw + THREADS - 1) / THREADS, THREADS, 0, stream>>>(wq, WqT, DMODEL, DMODEL);
    k_cvt_bf16_T<<<(nw + THREADS - 1) / THREADS, THREADS, 0, stream>>>(wk, WkT, DMODEL, DMODEL);
    k_cvt_bf16_T<<<(nw + THREADS - 1) / THREADS, THREADS, 0, stream>>>(wv, WvT, DMODEL, DMODEL);
  }

  // 1) Q/K/V projections: 3 * 256 * 16 strip-waves, 8 waves per block
  {
    int waves = 3 * (SEQ / 16) * (DMODEL / 64);
    k_proj<<<waves / 8, THREADS, 0, stream>>>(X1b, X2b, WqT, WkT, WvT, Qb, Kb, Vt);
  }

  // 2) softmax(Q K^T) -> P (one wave per 16 query rows)
  {
    int waves = SEQ / 16;
    k_softmax<<<waves / 8, THREADS, 0, stream>>>(Qb, Kb, P);
  }

  // 3) O = P V: 256 * 16 strip-waves
  {
    int waves = (SEQ / 16) * (DMODEL / 64);
    k_out<<<waves / 8, THREADS, 0, stream>>>(P, Vt, out);
  }
}
